// CrossModalHopfield_29635274342799
// MI455X (gfx1250) — matrix operand order repaired and checked by reading, not compiled, not run on hardware
//
#include <hip/hip_runtime.h>
#include <hip/hip_bf16.h>
#include <math.h>

typedef __bf16 bf16_t;
typedef __attribute__((ext_vector_type(16))) __bf16 v16bf;
typedef __attribute__((ext_vector_type(8)))  __bf16 v8bf;
typedef __attribute__((ext_vector_type(8)))  float  v8f;

// ---------------------------------------------------------------------------
// Stage 0: zero the scalar output (graph-replay safe; atomics accumulate later)
// ---------------------------------------------------------------------------
__global__ void cmh_zero_out(float* out) { out[0] = 0.0f; }

// ---------------------------------------------------------------------------
// Stage 1: bf16 cast + temporal smoothing (window=3, edge-truncated mean).
// f: (B*S, D) f32 -> fb: bf16 copy, sb: bf16 smoothed.  s = (idx/D) % S.
// ---------------------------------------------------------------------------
__global__ void cmh_smooth_cast(const float* __restrict__ f,
                                bf16_t* __restrict__ fb,
                                bf16_t* __restrict__ sb,
                                int S, int D, int total)
{
    int idx = blockIdx.x * blockDim.x + threadIdx.x;
    if (idx >= total) return;
    int s = (idx / D) % S;
    float c   = f[idx];
    float sum = c;
    float cnt = 1.0f;
    if (s > 0)     { sum += f[idx - D]; cnt += 1.0f; }
    if (s < S - 1) { sum += f[idx + D]; cnt += 1.0f; }
    fb[idx] = (bf16_t)c;
    sb[idx] = (bf16_t)(sum / cnt);
}

// ---------------------------------------------------------------------------
// Stage 2a: plain f32 -> bf16 cast (prototype matrices)
// ---------------------------------------------------------------------------
__global__ void cmh_cast_bf16(const float* __restrict__ src,
                              bf16_t* __restrict__ dst, int n)
{
    int i = blockIdx.x * blockDim.x + threadIdx.x;
    if (i < n) dst[i] = (bf16_t)src[i];
}

// ---------------------------------------------------------------------------
// Stage 2b: combined projection matrix  M[k][c] = sum_d proto[k][d] * W[d][c]
// proto: (K, Din) row-major, W: (Din, Cout) row-major. f32 accumulate -> bf16.
// Tiny (<=0.4 GFLOP each); adjacent threads share k and coalesce over c.
// ---------------------------------------------------------------------------
__global__ void cmh_combine(const float* __restrict__ proto,
                            const float* __restrict__ W,
                            bf16_t* __restrict__ out,
                            int Din, int Cout, int total)
{
    int i = blockIdx.x * blockDim.x + threadIdx.x;
    if (i >= total) return;
    int k = i / Cout;
    int c = i % Cout;
    const float* pr = proto + (size_t)k * Din;
    float acc = 0.0f;
    for (int d = 0; d < Din; ++d)
        acc = fmaf(pr[d], W[(size_t)d * Cout + c], acc);
    out[i] = (bf16_t)acc;
}

// ---------------------------------------------------------------------------
// Async global -> LDS copy of one 16-byte chunk (CDNA5 ASYNCcnt path).
// vdst = LDS byte address, vaddr = 64-bit global address, GV mode.
// ---------------------------------------------------------------------------
__device__ __forceinline__ void cmh_async_b128(unsigned lds_byte_off,
                                               const void* gaddr)
{
    asm volatile("global_load_async_to_lds_b128 %0, %1, off"
                 :: "v"(lds_byte_off), "v"(gaddr)
                 : "memory");
}

__device__ __forceinline__ void cmh_wait_async0()
{
    asm volatile("s_wait_asynccnt 0x0" ::: "memory");
}

// ---------------------------------------------------------------------------
// Stage 3: fused dual-GEMM + softplus-reduce energy kernel.
// One block = one 16-row s-tile of the flattened (B*S) axis.
// The block's feat tile (16 x Dm) and smoothed-comp tile (16 x Dc) are staged
// into LDS ONCE via async global->LDS B128 copies (row stride padded +8 bf16
// so the 16 A-lanes hit disjoint bank groups), then 8 waves x 4 k-tiles each
// run the K=512 prototype rows:
//   acc_i = feat_m @ proto_m^T  (A from LDS, B from global/L2)
//   acc_c = smooth_c @ Mcomb^T
// Epilogue: -softplus((lam*c^2 + (1-lam)*i^2) * i), block reduce, atomicAdd.
//
// A-fragment (16x32 bf16, M x K): lane L holds row L%16; chunks at d-offsets
//   (L/16)*16 .. +7  and  (L/16)*16+8 .. +15   (contiguous k, same order as B)
// B-fragment (32x16 bf16, K x N): lane L holds proto row kt+(L%16),
//   d-chunk (L/16)*16 .. +15  -> single contiguous 32B load (row-major proto).
// All copy loops have chunk counts that are exact multiples of 256 threads,
// so EXEC stays all-ones through every WMMA (ISA 7.12 requirement).
// ---------------------------------------------------------------------------
__global__ __launch_bounds__(256) void cmh_energy(
    const bf16_t* __restrict__ featM,    // (B*S, Dm)
    const bf16_t* __restrict__ protoM,   // (K,  Dm)
    const bf16_t* __restrict__ smoothC,  // (B*S, Dc)
    const bf16_t* __restrict__ Mcomb,    // (K,  Dc)
    int Dm, int Dc, float lam,
    float* __restrict__ out)
{
    extern __shared__ __align__(16) char smem[];

    const int tid  = threadIdx.x;
    const int lane = tid & 31;
    const int wave = tid >> 5;        // 0..7
    const int m    = lane & 15;
    const int hi   = lane >> 4;       // 0 or 1
    const int tile = blockIdx.x;

    const int LDAm = Dm + 8;          // padded LDS row strides (16B pad ->
    const int LDAc = Dc + 8;          //  4-bank rotation per row, no conflicts)
    bf16_t* ldsI = (bf16_t*)smem;                     // 16 x LDAm
    bf16_t* ldsC = (bf16_t*)smem + 16 * LDAm;         // 16 x LDAc
    float*  red  = (float*)((bf16_t*)smem + 16 * (LDAm + LDAc));

    const unsigned lbaseI =
        (unsigned)(size_t)(__attribute__((address_space(3))) char*)smem;
    const unsigned lbaseC = lbaseI + (unsigned)(16 * LDAm) * 2u;

    // ---- async stage: feat tile and smoothed-comp tile -> LDS ----
    {
        const bf16_t* gI = featM + (size_t)tile * 16 * Dm;
        const int cprI = Dm / 8;                    // 16B chunks per row
        for (int c = tid; c < 16 * cprI; c += 256) {  // 1536 or 1024: uniform
            int r = c / cprI, x = c - r * cprI;
            cmh_async_b128(lbaseI + (unsigned)(r * LDAm + x * 8) * 2u,
                           gI + (size_t)r * Dm + x * 8);
        }
        const bf16_t* gC = smoothC + (size_t)tile * 16 * Dc;
        const int cprC = Dc / 8;
        for (int c = tid; c < 16 * cprC; c += 256) {
            int r = c / cprC, x = c - r * cprC;
            cmh_async_b128(lbaseC + (unsigned)(r * LDAc + x * 8) * 2u,
                           gC + (size_t)r * Dc + x * 8);
        }
    }
    cmh_wait_async0();     // each wave drains its own ASYNCcnt
    __syncthreads();       // then LDS tile is visible workgroup-wide

    const bf16_t* arow_i = ldsI + m * LDAm;
    const bf16_t* arow_c = ldsC + m * LDAc;

    v8f acc_i[4], acc_c[4];
    v8f vzero = {};
#pragma unroll
    for (int j = 0; j < 4; ++j) { acc_i[j] = vzero; acc_c[j] = vzero; }

    // ---- intra GEMM: featM @ protoM^T over Dm (A from LDS) ----
    for (int d = 0; d < Dm; d += 32) {
        const bf16_t* pa = arow_i + d + hi * 16;
        v8bf lo = *(const v8bf*)(pa);
        v8bf hh = *(const v8bf*)(pa + 8);
        v16bf a;
#pragma unroll
        for (int i = 0; i < 8; ++i) { a[i] = lo[i]; a[i + 8] = hh[i]; }
#pragma unroll
        for (int j = 0; j < 4; ++j) {
            const int krow = (wave * 4 + j) * 16 + m;
            v16bf b = *(const v16bf*)(protoM + (size_t)krow * Dm + d + hi * 16);
            acc_i[j] = __builtin_amdgcn_wmma_f32_16x16x32_bf16(
                false, a, false, b, (short)0, acc_i[j], false, false);
        }
    }

    // ---- cross GEMM: smoothC @ Mcomb^T over Dc (A from LDS) ----
    for (int d = 0; d < Dc; d += 32) {
        const bf16_t* pa = arow_c + d + hi * 16;
        v8bf lo = *(const v8bf*)(pa);
        v8bf hh = *(const v8bf*)(pa + 8);
        v16bf a;
#pragma unroll
        for (int i = 0; i < 8; ++i) { a[i] = lo[i]; a[i + 8] = hh[i]; }
#pragma unroll
        for (int j = 0; j < 4; ++j) {
            const int krow = (wave * 4 + j) * 16 + m;
            v16bf b = *(const v16bf*)(Mcomb + (size_t)krow * Dc + d + hi * 16);
            acc_c[j] = __builtin_amdgcn_wmma_f32_16x16x32_bf16(
                false, a, false, b, (short)0, acc_c[j], false, false);
        }
    }

    // ---- fused epilogue: saliency -> softplus -> sum ----
    float local = 0.0f;
#pragma unroll
    for (int j = 0; j < 4; ++j) {
#pragma unroll
        for (int r = 0; r < 8; ++r) {
            float iv  = acc_i[j][r];
            float cv  = acc_c[j][r];
            float sal = lam * cv * cv + (1.0f - lam) * iv * iv;
            float x   = sal * iv;
            float sp  = (x > 15.0f) ? x : log1pf(__expf(x));
            local += sp;
        }
    }

    red[tid] = local;
    __syncthreads();
#pragma unroll
    for (int off = 128; off > 0; off >>= 1) {
        if (tid < off) red[tid] += red[tid + off];
        __syncthreads();
    }
    if (tid == 0) atomicAdd(out, -red[0]);
}

// ---------------------------------------------------------------------------
// Launcher
// ---------------------------------------------------------------------------
extern "C" void kernel_launch(void* const* d_in, const int* in_sizes, int n_in,
                              void* d_out, int out_size, void* d_ws, size_t ws_size,
                              hipStream_t stream)
{
    (void)in_sizes; (void)n_in; (void)out_size; (void)ws_size;

    const float* featV  = (const float*)d_in[0];   // (B,S,DV)
    const float* featA  = (const float*)d_in[1];   // (B,S,DA)
    const float* protoV = (const float*)d_in[2];   // (K,DV)
    const float* protoA = (const float*)d_in[3];   // (K,DA)
    const float* Wav    = (const float*)d_in[4];   // (DV,DA)
    const float* Wva    = (const float*)d_in[5];   // (DA,DV)
    float* out = (float*)d_out;

    const int B = 8, S = 2048, DV = 768, DA = 512, K = 512;
    const size_t NV = (size_t)B * S * DV;          // 12,582,912
    const size_t NA = (size_t)B * S * DA;          //  8,388,608

    // Workspace carve-up (all chunks are multiples of 256 bytes)
    char* ws = (char*)d_ws;
    bf16_t* featVb   = (bf16_t*)ws;  ws += NV * sizeof(bf16_t);
    bf16_t* featAb   = (bf16_t*)ws;  ws += NA * sizeof(bf16_t);
    bf16_t* smoothVb = (bf16_t*)ws;  ws += NV * sizeof(bf16_t);
    bf16_t* smoothAb = (bf16_t*)ws;  ws += NA * sizeof(bf16_t);
    bf16_t* protoVb  = (bf16_t*)ws;  ws += (size_t)K * DV * sizeof(bf16_t);
    bf16_t* protoAb  = (bf16_t*)ws;  ws += (size_t)K * DA * sizeof(bf16_t);
    bf16_t* Mv       = (bf16_t*)ws;  ws += (size_t)K * DA * sizeof(bf16_t);  // protoV @ Wav
    bf16_t* Ma       = (bf16_t*)ws;  ws += (size_t)K * DV * sizeof(bf16_t);  // protoA @ Wva

    cmh_zero_out<<<1, 1, 0, stream>>>(out);

    const int tpb = 256;
    cmh_smooth_cast<<<(int)((NV + tpb - 1) / tpb), tpb, 0, stream>>>(
        featV, featVb, smoothVb, S, DV, (int)NV);
    cmh_smooth_cast<<<(int)((NA + tpb - 1) / tpb), tpb, 0, stream>>>(
        featA, featAb, smoothAb, S, DA, (int)NA);

    cmh_cast_bf16<<<(K * DV + tpb - 1) / tpb, tpb, 0, stream>>>(protoV, protoVb, K * DV);
    cmh_cast_bf16<<<(K * DA + tpb - 1) / tpb, tpb, 0, stream>>>(protoA, protoAb, K * DA);

    // Mv[k][c] = sum_d protoV[k][d] * Wav[d][c]   (K x DA)
    cmh_combine<<<(K * DA + tpb - 1) / tpb, tpb, 0, stream>>>(
        protoV, Wav, Mv, DV, DA, K * DA);
    // Ma[k][c] = sum_d protoA[k][d] * Wva[d][c]   (K x DV)
    cmh_combine<<<(K * DV + tpb - 1) / tpb, tpb, 0, stream>>>(
        protoA, Wva, Ma, DA, DV, K * DV);

    // Dynamic LDS: 16 x (Dm+8) + 16 x (Dc+8) bf16 tiles + 256-float reduce.
    const size_t smem_bytes = (size_t)16 * (DV + 8) * 2 +
                              (size_t)16 * (DA + 8) * 2 + 256 * sizeof(float);

    const int tiles = B * S / 16;   // 1024 blocks
    // e_v: intra over DV, cross over DA
    cmh_energy<<<tiles, 256, smem_bytes, stream>>>(featVb, protoVb, smoothAb, Mv,
                                                   DV, DA, 0.3f, out);
    // e_a: intra over DA, cross over DV
    cmh_energy<<<tiles, 256, smem_bytes, stream>>>(featAb, protoAb, smoothVb, Ma,
                                                   DA, DV, 0.3f, out);
}
